// shift_Windows_56332791054554
// MI455X (gfx1250) — compile-verified, matrix operand order checked by reading, not run on hardware
//
#include <hip/hip_runtime.h>
#include <hip/hip_bf16.h>

typedef _Float16 v4h  __attribute__((ext_vector_type(4)));
typedef _Float16 v8h  __attribute__((ext_vector_type(8)));
typedef _Float16 v16h __attribute__((ext_vector_type(16)));
typedef float    v4f  __attribute__((ext_vector_type(4)));
typedef float    v8f  __attribute__((ext_vector_type(8)));

#define NPATCH 13824   // 24*24*24
#define LDS_STRIDE 40  // 32 halfs + 8 pad (80B rows, 16B aligned)

// Load one 16x32 f16 fragment (A or B operand) from LDS.
// Lane mapping per CDNA5 ISA: lane<16 -> rows 0-15 with K {h*8..h*8+7, 16+h*8..23+h*8}
__device__ __forceinline__ v16h load_frag(const _Float16* base, int row, int khalf) {
    const v8h lo = *(const v8h*)(base + row * LDS_STRIDE + khalf * 8);
    const v8h hi = *(const v8h*)(base + row * LDS_STRIDE + 16 + khalf * 8);
    return __builtin_shufflevector(lo, hi, 0,1,2,3,4,5,6,7,8,9,10,11,12,13,14,15);
}

__global__ __launch_bounds__(256, 2)
void swin_wmma_kernel(const float* __restrict__ x, const float* __restrict__ Wm,
                      const float* __restrict__ bias, float* __restrict__ out) {
    const int tid = threadIdx.x;
    const int p0 = blockIdx.x * 128;   // patch tile (stride-1 dim of x/out)
    const int n0 = blockIdx.y * 128;   // output-feature tile
    const int bg = blockIdx.z;
    const int b  = bg >> 1;
    const int g  = bg & 1;             // window index

    __shared__ __align__(16) _Float16 sA[128 * LDS_STRIDE];
    __shared__ __align__(16) _Float16 sB[128 * LDS_STRIDE];

    const int lane  = tid & 31;
    const int wv    = tid >> 5;
    const int wm    = wv >> 2;         // 0..1 : 64-row sub-block
    const int wn    = wv & 3;          // 0..3 : 32-col sub-block
    const int hrow  = lane & 15;
    const int khalf = lane >> 4;

    v8f acc[4][2];
    #pragma unroll
    for (int i = 0; i < 4; ++i)
        #pragma unroll
        for (int j = 0; j < 2; ++j) {
            v8f z = {0.f,0.f,0.f,0.f,0.f,0.f,0.f,0.f};
            acc[i][j] = z;
        }

    // A stage: 32 threads along i (x4 floats), 8 groups along k (x4 channels)
    const int ai = (tid & 31) * 4;
    const int ak = (tid >> 5) * 4;
    // B stage: 2 threads per W row, each covers 16 k (4x float4)
    const int bo = tid >> 1;
    const int bk = (tid & 1) * 16;

    #pragma unroll 1
    for (int ks = 0; ks < 32; ++ks) {
        // K-step of 32 == one 's' value; circular shift folds to +48 mod 64
        const int tgs = (g * 32 + ks + 48) & 63;

        v4f av[4], bv[4];
        #pragma unroll
        for (int q = 0; q < 4; ++q) {
            // x viewed as (B, C, 64, 13824); channel c = ak+q, contiguous in p
            av[q] = *(const v4f*)(x + (long)((b * 32 + ak + q) * 64 + tgs) * NPATCH + p0 + ai);
            // W row (n0+bo), K-contiguous
            bv[q] = *(const v4f*)(Wm + (long)(n0 + bo) * 1024 + ks * 32 + bk + q * 4);
        }

        __syncthreads();  // previous iteration's fragment reads done

        #pragma unroll
        for (int j = 0; j < 4; ++j) {   // transpose-in-registers: [k][i] -> [i][k]
            v4h pa = { (_Float16)av[0][j], (_Float16)av[1][j],
                       (_Float16)av[2][j], (_Float16)av[3][j] };
            *(v4h*)(sA + (ai + j) * LDS_STRIDE + ak) = pa;
        }
        #pragma unroll
        for (int q = 0; q < 4; ++q) {
            v4h pb = { (_Float16)bv[q][0], (_Float16)bv[q][1],
                       (_Float16)bv[q][2], (_Float16)bv[q][3] };
            *(v4h*)(sB + bo * LDS_STRIDE + bk + q * 4) = pb;
        }

        __syncthreads();

        v16h bf[2];
        #pragma unroll
        for (int fn = 0; fn < 2; ++fn)
            bf[fn] = load_frag(sB, wn * 32 + fn * 16 + hrow, khalf);

        #pragma unroll
        for (int fm = 0; fm < 4; ++fm) {
            v16h af = load_frag(sA, wm * 64 + fm * 16 + hrow, khalf);
            #pragma unroll
            for (int fn = 0; fn < 2; ++fn)
                acc[fm][fn] = __builtin_amdgcn_wmma_f32_16x16x32_f16(
                    /*neg_a=*/false, af, /*neg_b=*/false, bf[fn],
                    /*c_mod=*/(short)0, acc[fm][fn],
                    /*reuse_a=*/false, /*reuse_b=*/false);
        }
    }

    // Epilogue: D layout => vgpr v, lane L: m = v + 8*(L>>4) (+frag offs), n = L&15 (+frag offs)
    #pragma unroll
    for (int fn = 0; fn < 2; ++fn) {
        const int n  = n0 + wn * 32 + fn * 16 + hrow;     // output feature o
        const float bb = bias[n];
        const int cp = n & 31;                            // channel
        const int sp = n >> 5;                            // s'
        const int tp = (g * 32 + sp + 48) & 63;           // roll-back folds to same map
        float* orow = out + (long)((b * 32 + cp) * 64 + tp) * NPATCH + p0;
        #pragma unroll
        for (int fm = 0; fm < 4; ++fm) {
            const int m0 = wm * 64 + fm * 16 + khalf * 8; // p offset, 16B aligned
            v4f lo = { acc[fm][fn][0] + bb, acc[fm][fn][1] + bb,
                       acc[fm][fn][2] + bb, acc[fm][fn][3] + bb };
            v4f hi = { acc[fm][fn][4] + bb, acc[fm][fn][5] + bb,
                       acc[fm][fn][6] + bb, acc[fm][fn][7] + bb };
            *(v4f*)(orow + m0)     = lo;
            *(v4f*)(orow + m0 + 4) = hi;
        }
    }
}

extern "C" void kernel_launch(void* const* d_in, const int* in_sizes, int n_in,
                              void* d_out, int out_size, void* d_ws, size_t ws_size,
                              hipStream_t stream) {
    (void)in_sizes; (void)n_in; (void)d_ws; (void)ws_size; (void)out_size;
    const float* x  = (const float*)d_in[0];
    const float* Wm = (const float*)d_in[1];
    const float* bb = (const float*)d_in[2];
    float* out = (float*)d_out;

    dim3 grid(NPATCH / 128, 1024 / 128, 4);  // (108 p-tiles, 8 n-tiles, B*2 windows)
    dim3 block(256);
    swin_wmma_kernel<<<grid, block, 0, stream>>>(x, Wm, bb, out);
}
